// GPTNeoSelfAttention_22016002359998
// MI455X (gfx1250) — matrix operand order repaired and checked by reading, not compiled, not run on hardware
//
#include <hip/hip_runtime.h>
#include <hip/hip_bf16.h>
#include <stdint.h>

// GPT-Neo self-attention, MI455X (gfx1250), wave32 + WMMA bf16 path.
// B=2, S=2048, E=2048, H=16, D=128.
#define BB 2
#define SS 2048
#define EE 2048
#define HH 16
#define DD 128
#define MM (BB * SS) /* 4096 rows */

typedef __attribute__((ext_vector_type(16))) __bf16 v16bf;
typedef __attribute__((ext_vector_type(8))) float v8f;

// 16x32 bf16 fragment = 32 bytes per lane (two 16B chunks).
union Frag {
    v16bf v;
    uint4 q[2];
};

__device__ __forceinline__ unsigned short f2bf(float f) {
    unsigned u = __float_as_uint(f);
    u = (u + 0x7FFFu + ((u >> 16) & 1u)) >> 16; // round-to-nearest-even
    return (unsigned short)u;
}

// CDNA5 async Global->LDS copy (ASYNCcnt-tracked). Inline asm: portable across
// ROCm7.2 / amdgpu-toolchain builtin arity differences. VDST = LDS byte
// address (low 32 bits of the flat shared pointer), VADDR = 64-bit global.
__device__ __forceinline__ void async_copy_b128(const void* lds_dst,
                                                const void* gsrc) {
    unsigned lds_off = (unsigned)(uintptr_t)lds_dst;
    asm volatile("global_load_async_to_lds_b128 %0, %1, off" ::"v"(lds_off),
                 "v"(gsrc)
                 : "memory");
}

__global__ void convert_f32_to_bf16(const float* __restrict__ in,
                                    unsigned short* __restrict__ out, int n) {
    int i = blockIdx.x * blockDim.x + threadIdx.x;
    int stride = gridDim.x * blockDim.x;
    for (; i < n; i += stride) out[i] = f2bf(in[i]);
}

// C = A(M x K, bf16 row-major) * W^T, W is (N x K, bf16 row-major).
//   mode 0: out bf16, Q/K head layout [B,H,S,D]
//   mode 1: out bf16, V head-transposed layout [B,H,D,S]
//   mode 2: out fp32 [M,N] + bias[n]
// Block: 256 thr = 8 waves; wave = 32x32 tile; block tile = 64(M) x 128(N).
// A/B slabs double-buffered in LDS via async Global->LDS DMA so the copy for
// k+32 overlaps the WMMAs on k.
__global__ __launch_bounds__(256) void gemm_xwt_wmma(
    const unsigned short* __restrict__ A, const unsigned short* __restrict__ W,
    unsigned short* __restrict__ outb, float* __restrict__ outf,
    const float* __restrict__ bias, int mode) {
    __shared__ __align__(16) unsigned short As[2][64 * 32];  // 4KB x2
    __shared__ __align__(16) unsigned short Bs[2][128 * 32]; // 8KB x2

    const int K = EE;
    const int t = threadIdx.x;
    const int lane = t & 31;
    const int w = t >> 5;
    const int wy = w >> 2, wx = w & 3;
    const int m_blk = blockIdx.x * 64;
    const int n_blk = blockIdx.y * 128;
    const int l15 = lane & 15;
    const int lh = lane >> 4;

    v8f acc[2][2] = {};

    // Per-thread async-copy assignments (3 x b128 per buffer fill):
    //   A slab 64x32 = 256 16B chunks -> 1/thread; B slab 128x32 = 512 -> 2.
    const int a_row = t >> 2, a_kk = (t & 3) * 8;
    auto issue = [&](int buf, int k0) {
        async_copy_b128(&As[buf][a_row * 32 + a_kk],
                        A + (size_t)(m_blk + a_row) * K + k0 + a_kk);
#pragma unroll
        for (int j = 0; j < 2; ++j) {
            int c = t + 256 * j;
            int col = c >> 2, kk = (c & 3) * 8;
            async_copy_b128(&Bs[buf][col * 32 + kk],
                            W + (size_t)(n_blk + col) * K + k0 + kk);
        }
    };

    issue(0, 0);
    int buf = 0;
    for (int k0 = 0; k0 < K; k0 += 32) {
        const bool has_next = (k0 + 32) < K;
        if (has_next) {
            issue(buf ^ 1, k0 + 32);
            asm volatile("s_wait_asynccnt 0x3" ::: "memory"); // cur 3 done
        } else {
            asm volatile("s_wait_asynccnt 0x0" ::: "memory");
        }
        __syncthreads(); // all waves' async fills visible

        // A-layout (16x32 bf16): lane l15 = row; halves 0-7 -> K lh*8..+7,
        // halves 8-15 -> K lh*8+16..+23. B-layout (32x16): lane l15 = col,
        // same K order as A: halves 0-7 -> K lh*8..+7, 8-15 -> K lh*8+16..+23.
        Frag a[2], b[2];
#pragma unroll
        for (int i = 0; i < 2; ++i) {
            const unsigned short* ar = &As[buf][(32 * wy + 16 * i + l15) * 32];
            a[i].q[0] = *(const uint4*)(ar + lh * 8);
            a[i].q[1] = *(const uint4*)(ar + lh * 8 + 16);
            const unsigned short* br = &Bs[buf][(32 * wx + 16 * i + l15) * 32];
            b[i].q[0] = *(const uint4*)(br + lh * 8);
            b[i].q[1] = *(const uint4*)(br + lh * 8 + 16);
        }
#pragma unroll
        for (int i = 0; i < 2; ++i)
#pragma unroll
            for (int j = 0; j < 2; ++j)
                acc[i][j] = __builtin_amdgcn_wmma_f32_16x16x32_bf16(
                    false, a[i].v, false, b[j].v, (short)0, acc[i][j], false,
                    false);
        __syncthreads(); // reads done before this buffer is refilled
        buf ^= 1;
    }

    // C-layout: vgpr r, lanes 0-15 -> row r, lanes 16-31 -> row r+8; col = l15.
#pragma unroll
    for (int i = 0; i < 2; ++i)
#pragma unroll
        for (int j = 0; j < 2; ++j)
#pragma unroll
            for (int r = 0; r < 8; ++r) {
                int m = m_blk + 32 * wy + 16 * i + r + 8 * lh;
                int n = n_blk + 32 * wx + 16 * j + l15;
                float val = acc[i][j][r];
                if (mode == 2) {
                    outf[(size_t)m * EE + n] = val + bias[n];
                } else {
                    int bi = m >> 11, s = m & (SS - 1);
                    int h = n >> 7, d = n & (DD - 1);
                    size_t idx =
                        (mode == 0)
                            ? ((size_t)(bi * HH + h) * SS + s) * DD + d
                            : ((size_t)(bi * HH + h) * DD + d) * SS + s;
                    outb[idx] = f2bf(val);
                }
            }
}

// Flash attention with online softmax. Q,K: [B,H,S,D] bf16; Vt: [B,H,D,S] bf16.
// Output attn: [B,S,H*D] bf16. Block = 128 thr = 4 waves; each wave owns 16
// query rows; grid = (S/64, B*H). GPT-Neo: no 1/sqrt(d) scaling, causal mask
// bias = -1e9, softmax in fp32.
__global__ __launch_bounds__(128) void flash_attn_wmma(
    const unsigned short* __restrict__ Q, const unsigned short* __restrict__ Km,
    const unsigned short* __restrict__ Vt, unsigned short* __restrict__ attb) {
    __shared__ __align__(16) unsigned short Plds[4][16][40]; // per-wave slabs
    const int lane = threadIdx.x & 31;
    const int w = threadIdx.x >> 5; // 0..3
    const int l15 = lane & 15, lh = lane >> 4;
    const int q0 = blockIdx.x * 64;
    const int bh = blockIdx.y;
    const int qw = q0 + w * 16;

    const size_t headQK = (size_t)bh * SS * DD;
    const size_t headV = (size_t)bh * DD * SS;

    // Q fragments: 16 rows x 128 d = 4 A-frags (16x32), loaded once.
    Frag qf[4];
    {
        const unsigned short* qrow = Q + headQK + (size_t)(qw + l15) * DD;
        const int kb = lh * 8;
#pragma unroll
        for (int j = 0; j < 4; ++j) {
            qf[j].q[0] = *(const uint4*)(qrow + 32 * j + kb);
            qf[j].q[1] = *(const uint4*)(qrow + 32 * j + kb + 16);
        }
    }

    v8f o[8] = {}; // 16 x 128 fp32 output accumulator
    float mi[8], li[8];
#pragma unroll
    for (int r = 0; r < 8; ++r) {
        mi[r] = -1e30f;
        li[r] = 0.0f;
    }

    const int nsteps = q0 / 32 + 2; // causal: keys [0, q0+64)
    for (int step = 0; step < nsteps; ++step) {
        const int kv0 = step * 32;
        // scores S = Q * K^T : two 16x16 f32 tiles (keys kv0..+15, +16..+31)
        v8f sc[2] = {};
        {
            const int kb = lh * 8;
#pragma unroll
            for (int t = 0; t < 2; ++t) {
                const unsigned short* krow =
                    Km + headQK + (size_t)(kv0 + 16 * t + l15) * DD + kb;
                __builtin_prefetch(krow + 32 * DD, 0, 0);
#pragma unroll
                for (int j = 0; j < 4; ++j) {
                    Frag kf;
                    kf.q[0] = *(const uint4*)(krow + 32 * j);
                    kf.q[1] = *(const uint4*)(krow + 32 * j + 16);
                    sc[t] = __builtin_amdgcn_wmma_f32_16x16x32_bf16(
                        false, qf[j].v, false, kf.v, (short)0, sc[t], false,
                        false);
                }
            }
        }

        // causal mask + online softmax (row = r + 8*lh across 16-lane groups)
        float p0[8], p1[8];
#pragma unroll
        for (int r = 0; r < 8; ++r) {
            int row = qw + r + 8 * lh;
            float s0 = (kv0 + l15 <= row) ? sc[0][r] : -1e9f;
            float s1 = (kv0 + 16 + l15 <= row) ? sc[1][r] : -1e9f;
            float t = fmaxf(s0, s1);
#pragma unroll
            for (int msk = 1; msk < 16; msk <<= 1)
                t = fmaxf(t, __shfl_xor(t, msk, 32));
            float mnew = fmaxf(mi[r], t);
            float scale = __expf(mi[r] - mnew);
            float e0 = __expf(s0 - mnew);
            float e1 = __expf(s1 - mnew);
            float rs = e0 + e1;
#pragma unroll
            for (int msk = 1; msk < 16; msk <<= 1) rs += __shfl_xor(rs, msk, 32);
            li[r] = li[r] * scale + rs;
            mi[r] = mnew;
            p0[r] = e0;
            p1[r] = e1;
#pragma unroll
            for (int t8 = 0; t8 < 8; ++t8) o[t8][r] *= scale;
        }

        // C-layout -> A-layout for P via this wave's private LDS slab.
        // In-wave DS ops complete in order; no cross-wave sharing -> no barrier.
#pragma unroll
        for (int r = 0; r < 8; ++r) {
            int row = r + 8 * lh;
            Plds[w][row][l15] = f2bf(p0[r]);
            Plds[w][row][16 + l15] = f2bf(p1[r]);
        }
        Frag pf;
        {
            const int kb = lh * 8;
            pf.q[0] = *(const uint4*)&Plds[w][l15][kb];
            pf.q[1] = *(const uint4*)&Plds[w][l15][kb + 16];
        }

        // o += P (16x32) x V (32x128): Vt column n is contiguous in memory.
#pragma unroll
        for (int t = 0; t < 8; ++t) {
            const unsigned short* vcol =
                Vt + headV + (size_t)(16 * t + l15) * SS + kv0 + lh * 8;
            Frag vf;
            vf.q[0] = *(const uint4*)(vcol);
            vf.q[1] = *(const uint4*)(vcol + 16);
            o[t] = __builtin_amdgcn_wmma_f32_16x16x32_bf16(
                false, pf.v, false, vf.v, (short)0, o[t], false, false);
        }
    }

    // normalize + write attn [B,S,H*D] bf16 (row-major for the O-projection)
    const int b = bh >> 4, h = bh & 15;
#pragma unroll
    for (int t = 0; t < 8; ++t)
#pragma unroll
        for (int r = 0; r < 8; ++r) {
            int row = qw + r + 8 * lh;
            int d = 16 * t + l15;
            float val = o[t][r] / li[r];
            attb[((size_t)(b * SS + row)) * EE + h * DD + d] = f2bf(val);
        }
}

extern "C" void kernel_launch(void* const* d_in, const int* in_sizes, int n_in,
                              void* d_out, int out_size, void* d_ws,
                              size_t ws_size, hipStream_t stream) {
    const float* x = (const float*)d_in[0];
    const float* wq = (const float*)d_in[1];
    const float* wk = (const float*)d_in[2];
    const float* wv = (const float*)d_in[3];
    const float* wo = (const float*)d_in[4];
    const float* bo = (const float*)d_in[5];
    unsigned short* ws = (unsigned short*)d_ws;

    const size_t nX = (size_t)MM * EE; // 8,388,608
    const size_t nW = (size_t)EE * EE; // 4,194,304
    // Workspace layout (bf16 elements), ~112 MB total:
    unsigned short* Xb = ws;
    unsigned short* Wqb = Xb + nX;
    unsigned short* Wkb = Wqb + nW;
    unsigned short* Wvb = Wkb + nW;
    unsigned short* Wob = Wvb + nW;
    unsigned short* Qb = Wob + nW;  // [B,H,S,D]
    unsigned short* Kb = Qb + nX;   // [B,H,S,D]
    unsigned short* Vtb = Kb + nX;  // [B,H,D,S]
    unsigned short* Atb = Vtb + nX; // [B,S,E]

    convert_f32_to_bf16<<<2048, 256, 0, stream>>>(x, Xb, (int)nX);
    convert_f32_to_bf16<<<1024, 256, 0, stream>>>(wq, Wqb, (int)nW);
    convert_f32_to_bf16<<<1024, 256, 0, stream>>>(wk, Wkb, (int)nW);
    convert_f32_to_bf16<<<1024, 256, 0, stream>>>(wv, Wvb, (int)nW);
    convert_f32_to_bf16<<<1024, 256, 0, stream>>>(wo, Wob, (int)nW);

    dim3 gg(MM / 64, EE / 128); // 64 x 16 blocks
    gemm_xwt_wmma<<<gg, 256, 0, stream>>>(Xb, Wqb, Qb, nullptr, nullptr, 0);
    gemm_xwt_wmma<<<gg, 256, 0, stream>>>(Xb, Wkb, Kb, nullptr, nullptr, 0);
    gemm_xwt_wmma<<<gg, 256, 0, stream>>>(Xb, Wvb, Vtb, nullptr, nullptr, 1);

    dim3 ga(SS / 64, BB * HH); // 32 x 32 blocks
    flash_attn_wmma<<<ga, 128, 0, stream>>>(Qb, Kb, Vtb, Atb);

    gemm_xwt_wmma<<<gg, 256, 0, stream>>>(Atb, Wob, nullptr, (float*)d_out, bo,
                                          2);
}
